// RopeAttention_49331994362394
// MI455X (gfx1250) — compile-verified, matrix operand order checked by reading, not run on hardware
//
#include <hip/hip_runtime.h>

// ---------------------------------------------------------------------------
// RoPE causal attention block for MI455X (gfx1250, wave32, WMMA bf16).
//   B=2, T=2048, DIM=2048, H=16, HD=128
// Pipeline: fp32->bf16 convert -> qkv GEMM (WMMA, async-LDS double buffer)
//           -> RoPE split -> flash attention (WMMA, async-LDS double buffer)
//           -> out GEMM (WMMA)
// ---------------------------------------------------------------------------

#define BATCH 2
#define SEQ 2048
#define DIM 2048
#define NHEAD 16
#define HDIM 128

typedef __attribute__((ext_vector_type(4)))  float  v4f;
typedef __attribute__((ext_vector_type(4)))  __bf16 v4bf;
typedef __attribute__((ext_vector_type(8)))  __bf16 v8bf;
typedef __attribute__((ext_vector_type(16))) __bf16 v16bf;
typedef __attribute__((ext_vector_type(8)))  float  v8f;
typedef __attribute__((ext_vector_type(4)))  int    int4v;

// gfx1250 async global->LDS copies (ASYNCcnt) when the toolchain exposes them.
#if defined(__AMDGCN__) && __has_builtin(__builtin_amdgcn_global_load_async_to_lds_b128)
#define HAVE_ASYNC 1
typedef __attribute__((address_space(1))) int4v as1_int4;
typedef __attribute__((address_space(3))) int4v as3_int4;
#else
#define HAVE_ASYNC 0
#endif

// Copy 16 bytes (8 bf16) global -> LDS.
// AMDGCN ABI: generic global ptr value == AS1 address; generic LDS ptr is
// {aperture, offset} with offset in the low 32 bits == AS3 pointer value.
__device__ __forceinline__ void cp16(__bf16* l, const __bf16* g) {
#if HAVE_ASYNC
  as1_int4* gp = (as1_int4*)(unsigned long long)g;
  as3_int4* lp = (as3_int4*)(unsigned int)(unsigned long long)l;
  __builtin_amdgcn_global_load_async_to_lds_b128(gp, lp, 0, 0);
#else
  *(v8bf*)l = *(const v8bf*)g;
#endif
}

__device__ __forceinline__ void cp_wait() {
#if HAVE_ASYNC
#if __has_builtin(__builtin_amdgcn_s_wait_asynccnt)
  __builtin_amdgcn_s_wait_asynccnt(0);
#else
  asm volatile("s_wait_asynccnt 0" ::: "memory");
#endif
#endif
}

// Load a 16x32 bf16 fragment (A layout; B layout is the mirror with N per
// lane). CDNA5 16-bit operand layout:
//   lanes 0-15 : row = lane,     K = {0..7, 16..23}
//   lanes 16-31: row = lane-16,  K = {8..15, 24..31}
__device__ __forceinline__ v16bf load_frag(const __bf16* p, int ld) {
  const int lane = threadIdx.x & 31;
  const int r    = lane & 15;
  const int kb   = (lane >> 4) << 3;  // 0 or 8
  const __bf16* q = p + r * ld + kb;
  v8bf lo = *(const v8bf*)(q);
  v8bf hi = *(const v8bf*)(q + 16);
  v16bf f;
#pragma unroll
  for (int i = 0; i < 8; ++i) { f[i] = lo[i]; f[8 + i] = hi[i]; }
  return f;
}

// ---------------------------------------------------------------------------
// fp32 -> bf16 bulk convert (one-shot; lets all GEMM staging be byte copies)
// ---------------------------------------------------------------------------
__global__ void __launch_bounds__(256)
cvt_bf16(const float* __restrict__ in, __bf16* __restrict__ out) {
  const size_t i = ((size_t)blockIdx.x * 256 + threadIdx.x) * 4;
  const v4f v = *(const v4f*)(in + i);
  v4bf o;
#pragma unroll
  for (int j = 0; j < 4; ++j) o[j] = (__bf16)v[j];
  *(v4bf*)(out + i) = o;
}

// ---------------------------------------------------------------------------
// C[M,N] = A[M,K] * W[N,K]^T  (bf16 in, fp32 out). Block = 256 thr (8 waves),
// 128x128 tile, wave tile 32x64 -> 8 WMMAs per 6 fragment loads per K-step.
// LDS ping-pong + async copies overlap global traffic with WMMA.
// ---------------------------------------------------------------------------
__device__ __forceinline__ void stage_tile(const __bf16* __restrict__ A,
                                           const __bf16* __restrict__ W,
                                           __bf16* as, __bf16* bs,
                                           int mb, int nb, int K, int kk) {
  const int row = threadIdx.x >> 1;          // 0..127
  const int ko  = (threadIdx.x & 1) * 16;    // 0 or 16
  const __bf16* ga = A + (size_t)(mb + row) * K + kk + ko;
  const __bf16* gb = W + (size_t)(nb + row) * K + kk + ko;
  cp16(&as[row * 32 + ko],     ga);
  cp16(&as[row * 32 + ko + 8], ga + 8);
  cp16(&bs[row * 32 + ko],     gb);
  cp16(&bs[row * 32 + ko + 8], gb + 8);
}

__global__ void __launch_bounds__(256)
gemm_bb(const __bf16* __restrict__ A, const __bf16* __restrict__ W,
        float* __restrict__ C, int M, int N, int K) {
  __shared__ __attribute__((aligned(16))) __bf16 As[2][128 * 32];
  __shared__ __attribute__((aligned(16))) __bf16 Bs[2][128 * 32];

  const int mb = blockIdx.y * 128;
  const int nb = blockIdx.x * 128;
  const int w  = threadIdx.x >> 5;
  const int m_off = (w & 3) * 32;   // 4 waves over M
  const int n_off = (w >> 2) * 64;  // 2 waves over N

  v8f acc[2][4] = {};

  const int nk = K >> 5;
  stage_tile(A, W, As[0], Bs[0], mb, nb, K, 0);

  for (int i = 0; i < nk; ++i) {
    cp_wait();
    __syncthreads();
    if (i + 1 < nk)
      stage_tile(A, W, As[(i + 1) & 1], Bs[(i + 1) & 1], mb, nb, K, (i + 1) * 32);

    const __bf16* as = As[i & 1];
    const __bf16* bs = Bs[i & 1];
    v16bf a0 = load_frag(&as[(m_off) * 32], 32);
    v16bf a1 = load_frag(&as[(m_off + 16) * 32], 32);
    v16bf b[4];
#pragma unroll
    for (int nt = 0; nt < 4; ++nt)
      b[nt] = load_frag(&bs[(n_off + nt * 16) * 32], 32);
#pragma unroll
    for (int nt = 0; nt < 4; ++nt) {
      acc[0][nt] = __builtin_amdgcn_wmma_f32_16x16x32_bf16(
          false, a0, false, b[nt], (short)0, acc[0][nt], false, false);
      acc[1][nt] = __builtin_amdgcn_wmma_f32_16x16x32_bf16(
          false, a1, false, b[nt], (short)0, acc[1][nt], false, false);
    }
  }

  const int lane = threadIdx.x & 31;
  const int n  = lane & 15;
  const int m0 = (lane >> 4) << 3;
#pragma unroll
  for (int mt = 0; mt < 2; ++mt)
#pragma unroll
    for (int nt = 0; nt < 4; ++nt) {
#pragma unroll
      for (int r = 0; r < 8; ++r) {
        const size_t row = (size_t)(mb + m_off + mt * 16 + m0 + r);
        C[row * N + nb + n_off + nt * 16 + n] = acc[mt][nt][r];
      }
    }
}

// ---------------------------------------------------------------------------
// RoPE + split: qkv fp32 (B*T, 3*DIM) ->
//   Q,K bf16 (B,H,T,HD) (rope, reference even/odd interleave)
//   Vt  bf16 (B,H,HD,T) (transposed: attention PV B-frags contiguous)
// ---------------------------------------------------------------------------
__global__ void __launch_bounds__(256)
rope_split(const float* __restrict__ qkv, __bf16* __restrict__ Qb,
           __bf16* __restrict__ Kb, __bf16* __restrict__ Vt) {
  const int tid = blockIdx.x * blockDim.x + threadIdx.x;
  const int col = tid & (DIM - 1);
  const int t   = (tid >> 11) & (SEQ - 1);
  const int b   = tid >> 22;
  const int h   = col >> 7;
  const int d   = col & 127;
  const int half = d & 63;

  const float inv = __expf(-(float)half * 0.14391156f);  // ln(1e4)/64
  float s, c;
  __sincosf((float)t * inv, &s, &c);

  const float* row = qkv + (size_t)((b << 11) + t) * (3 * DIM);
  const int base = (h << 7) + (half << 1);
  const float q1 = row[base],       q2 = row[base + 1];
  const float k1 = row[DIM + base], k2 = row[DIM + base + 1];
  const float qo = (d < 64) ? (q1 * c - q2 * s) : (q2 * c + q1 * s);
  const float ko = (d < 64) ? (k1 * c - k2 * s) : (k2 * c + k1 * s);
  const float vo = row[2 * DIM + (h << 7) + d];

  const size_t bh = (size_t)(b * NHEAD + h);
  Qb[(bh * SEQ + t) * HDIM + d] = (__bf16)qo;
  Kb[(bh * SEQ + t) * HDIM + d] = (__bf16)ko;
  Vt[(bh * HDIM + d) * SEQ + t] = (__bf16)vo;
}

// ---------------------------------------------------------------------------
// Flash attention (causal, online softmax). Block = 128 thr (4 waves), each
// wave owns 16 q rows (64-row q tile per block); key tile = 32, double
// buffered K/V in LDS via async copies. grid = (T/64, H, B).
// ---------------------------------------------------------------------------
__device__ __forceinline__ void stage_kv(const __bf16* __restrict__ Kg,
                                         const __bf16* __restrict__ Vg,
                                         __bf16* ks, __bf16* vs,
                                         size_t bh, int j) {
  const int key = threadIdx.x >> 2;
  const int dc  = (threadIdx.x & 3) * 32;
  const __bf16* kg = Kg + (bh * SEQ + j + key) * HDIM + dc;
  __bf16* kd = &ks[key * 128 + dc];
#pragma unroll
  for (int i = 0; i < 4; ++i) cp16(kd + i * 8, kg + i * 8);
  const __bf16* vg = Vg + (bh * HDIM + threadIdx.x) * SEQ + j;
  __bf16* vd = &vs[threadIdx.x * 32];
#pragma unroll
  for (int i = 0; i < 4; ++i) cp16(vd + i * 8, vg + i * 8);
}

__global__ void __launch_bounds__(128)
flash_attn(const __bf16* __restrict__ Q, const __bf16* __restrict__ K,
           const __bf16* __restrict__ Vt, __bf16* __restrict__ O) {
  __shared__ __attribute__((aligned(16))) __bf16 Ks[2][32 * 128];  // [key][dim]
  __shared__ __attribute__((aligned(16))) __bf16 Vs[2][128 * 32];  // [dim][key]
  __shared__ __attribute__((aligned(16))) __bf16 Ps[4][16 * 32];   // per-wave P

  const int wv   = threadIdx.x >> 5;
  const int lane = threadIdx.x & 31;
  const int h = blockIdx.y, b = blockIdx.z;
  const size_t bh = (size_t)(b * NHEAD + h);
  const int q0 = blockIdx.x * 64 + wv * 16;

  v16bf qf[4];
  const __bf16* qbase = Q + (bh * SEQ + q0) * HDIM;
#pragma unroll
  for (int kc = 0; kc < 4; ++kc) qf[kc] = load_frag(qbase + kc * 32, HDIM);

  v8f o[8] = {};
  float mrow[8], lrow[8];
#pragma unroll
  for (int r = 0; r < 8; ++r) { mrow[r] = -3.0e38f; lrow[r] = 0.0f; }

  const int n  = lane & 15;
  const int m0 = (lane >> 4) << 3;
  const int njt = (blockIdx.x + 1) * 2;  // key tiles of 32 up to block end
  const float scale = 0.08838834764831845f;  // 1/sqrt(128)

  stage_kv(K, Vt, Ks[0], Vs[0], bh, 0);

  for (int jj = 0; jj < njt; ++jj) {
    cp_wait();
    __syncthreads();
    if (jj + 1 < njt)
      stage_kv(K, Vt, Ks[(jj + 1) & 1], Vs[(jj + 1) & 1], bh, (jj + 1) * 32);

    const int j = jj * 32;
    const __bf16* ksb = Ks[jj & 1];
    const __bf16* vsb = Vs[jj & 1];

    // S(16x32) = Q(16x128) . K^T
    v8f sc[2];
#pragma unroll
    for (int nt = 0; nt < 2; ++nt) {
      v8f a = {};
#pragma unroll
      for (int kc = 0; kc < 4; ++kc) {
        v16bf kf = load_frag(&ksb[(nt * 16) * 128 + kc * 32], 128);
        a = __builtin_amdgcn_wmma_f32_16x16x32_bf16(false, qf[kc], false, kf,
                                                    (short)0, a, false, false);
      }
      sc[nt] = a;
    }

    // scale + causal mask + online softmax (row stats via width-16 shfl_xor)
    float tmax[8];
#pragma unroll
    for (int r = 0; r < 8; ++r) {
      const int qi = q0 + m0 + r;
#pragma unroll
      for (int nt = 0; nt < 2; ++nt) {
        float sv = sc[nt][r] * scale;
        if (j + nt * 16 + n > qi) sv = -3.0e38f;
        sc[nt][r] = sv;
      }
      float v = fmaxf(sc[0][r], sc[1][r]);
#pragma unroll
      for (int off = 8; off > 0; off >>= 1) v = fmaxf(v, __shfl_xor(v, off, 32));
      tmax[r] = v;
    }
#pragma unroll
    for (int r = 0; r < 8; ++r) {
      const float mnew  = fmaxf(mrow[r], tmax[r]);
      const float alpha = __expf(mrow[r] - mnew);
      mrow[r] = mnew;
      const float p0 = __expf(sc[0][r] - mnew);
      const float p1 = __expf(sc[1][r] - mnew);
      sc[0][r] = p0;
      sc[1][r] = p1;
      float rs = p0 + p1;
#pragma unroll
      for (int off = 8; off > 0; off >>= 1) rs += __shfl_xor(rs, off, 32);
      lrow[r] = lrow[r] * alpha + rs;
#pragma unroll
      for (int d = 0; d < 8; ++d) o[d][r] *= alpha;
    }

    // C-layout fp32 P -> A-layout bf16 via per-wave LDS bounce
#pragma unroll
    for (int r = 0; r < 8; ++r) {
      Ps[wv][(m0 + r) * 32 + n]      = (__bf16)sc[0][r];
      Ps[wv][(m0 + r) * 32 + 16 + n] = (__bf16)sc[1][r];
    }
    __syncthreads();

    v16bf pf = load_frag(&Ps[wv][0], 32);
#pragma unroll
    for (int d = 0; d < 8; ++d) {
      v16bf vf = load_frag(&vsb[(d * 16) * 32], 32);
      o[d] = __builtin_amdgcn_wmma_f32_16x16x32_bf16(false, pf, false, vf,
                                                     (short)0, o[d], false, false);
    }
  }

  // epilogue: normalize, store bf16 (B,T,DIM) with head h at cols h*128
#pragma unroll
  for (int r = 0; r < 8; ++r) {
    const float rl = 1.0f / lrow[r];
    const size_t row = (size_t)b * SEQ + (q0 + m0 + r);
#pragma unroll
    for (int d = 0; d < 8; ++d)
      O[row * DIM + h * HDIM + d * 16 + n] = (__bf16)(o[d][r] * rl);
  }
}

// ---------------------------------------------------------------------------
extern "C" void kernel_launch(void* const* d_in, const int* in_sizes, int n_in,
                              void* d_out, int out_size, void* d_ws,
                              size_t ws_size, hipStream_t stream) {
  (void)in_sizes; (void)n_in; (void)out_size; (void)ws_size;

  const float* x    = (const float*)d_in[0];   // (B,T,DIM) fp32
  const float* Wqkv = (const float*)d_in[1];   // (3*DIM, DIM) fp32
  const float* Wout = (const float*)d_in[2];   // (DIM, DIM) fp32
  float*       out  = (float*)d_out;           // (B,T,DIM) fp32

  const int M = BATCH * SEQ;  // 4096
  char* ws = (char*)d_ws;
  size_t off = 0;
  float* qkv = (float*)(ws + off);             // (M, 3*DIM) fp32, ~100.7 MB
  off += (size_t)M * 3 * DIM * sizeof(float);
  __bf16* Qb = (__bf16*)(ws + off);            // (B,H,T,HD) bf16
  off += (size_t)BATCH * NHEAD * SEQ * HDIM * sizeof(__bf16);
  __bf16* Kb = (__bf16*)(ws + off);
  off += (size_t)BATCH * NHEAD * SEQ * HDIM * sizeof(__bf16);
  __bf16* Vt = (__bf16*)(ws + off);            // (B,H,HD,T) bf16
  off += (size_t)BATCH * NHEAD * SEQ * HDIM * sizeof(__bf16);
  __bf16* xb = (__bf16*)(ws + off);            // bf16 copies of inputs
  off += (size_t)M * DIM * sizeof(__bf16);
  __bf16* Wqkvb = (__bf16*)(ws + off);
  off += (size_t)3 * DIM * DIM * sizeof(__bf16);
  __bf16* Woutb = (__bf16*)(ws + off);
  off += (size_t)DIM * DIM * sizeof(__bf16);
  __bf16* attn = (__bf16*)ws;                  // alias qkv (dead after rope)

  // 0) one-shot fp32 -> bf16 conversion of GEMM operands
  cvt_bf16<<<(M * DIM) / 1024, 256, 0, stream>>>(x, xb);
  cvt_bf16<<<(3 * DIM * DIM) / 1024, 256, 0, stream>>>(Wqkv, Wqkvb);
  cvt_bf16<<<(DIM * DIM) / 1024, 256, 0, stream>>>(Wout, Woutb);

  // 1) qkv = x @ Wqkv^T   (M=4096, N=6144, K=2048)
  gemm_bb<<<dim3((3 * DIM) / 128, M / 128), 256, 0, stream>>>(
      xb, Wqkvb, qkv, M, 3 * DIM, DIM);

  // 2) RoPE + split into Q, K, V^T (bf16)
  rope_split<<<(BATCH * SEQ * DIM) / 256, 256, 0, stream>>>(qkv, Qb, Kb, Vt);

  // 3) causal flash attention -> attn bf16 (B,T,DIM)
  flash_attn<<<dim3(SEQ / 64, NHEAD, BATCH), 128, 0, stream>>>(Qb, Kb, Vt, attn);

  // 4) out = attn @ Wout^T  (M=4096, N=2048, K=2048)
  gemm_bb<<<dim3(DIM / 128, M / 128), 256, 0, stream>>>(
      attn, Woutb, out, M, DIM, DIM);
}
